// LVCNetGenerator_23991687316185
// MI455X (gfx1250) — matrix operand order repaired and checked by reading, compiled, not run on hardware
//
#include <hip/hip_runtime.h>
#include <hip/hip_fp16.h>

typedef __attribute__((ext_vector_type(16))) _Float16 v16h;
typedef __attribute__((ext_vector_type(8)))  _Float16 v8h;
typedef __attribute__((ext_vector_type(8)))  float    v8f;

#define B_      2
#define L_      64
#define HOP_    256
#define T_      (L_*HOP_)          // 16384
#define IN_CH_  101
#define INNER_  128
#define COND_   81
#define NBLOCK_ 2
#define LAYERS_ 5
#define K_      3
#define KH_     64
#define COUT_   (2*INNER_)         // 256
#define KCH_    (INNER_*COUT_*K_*LAYERS_)  // 491520
#define BCH_    (COUT_*LAYERS_)            // 1280
#define MROWS_  (INNER_*COUT_*K_)          // 98304 rows per layer
#define KD_     (INNER_*K_)                // 384 contraction dim per window
#define KDP_    392                        // LDS-padded A row stride (bank-conflict-free)
#define XS2_    136                        // LDS-padded x row stride  (bank-conflict-free)
#define XROWS_  288                        // max window length (256 + 2*16)

// async global->LDS copy path (gfx1250); compile-guarded so absence is harmless
#if defined(__has_builtin)
#  if __has_builtin(__builtin_amdgcn_global_load_async_to_lds_b128) && \
      __has_builtin(__builtin_amdgcn_s_wait_asynccnt)
#    define USE_ASYNC_LDS 1
#  endif
#endif
#ifndef USE_ASYNC_LDS
#  define USE_ASYNC_LDS 0
#endif

#if USE_ASYNC_LDS
typedef int async_v4i __attribute__((vector_size(16)));   // GCC-vector int4: builtin's pointee type
#define ASYNC_GV4(p) ((__attribute__((address_space(1))) async_v4i*)(void*)(p))
#define ASYNC_LV4(p) ((__attribute__((address_space(3))) async_v4i*)(void*)(p))
#endif

// ---------------------------------------------------------------- fc 1x1 conv
__global__ __launch_bounds__(256) void fc_kernel(
    const float* __restrict__ x, const float* __restrict__ W,
    const float* __restrict__ bias, float* __restrict__ h) {
  __shared__ float Wl[INNER_*IN_CH_];
  for (int i = threadIdx.x; i < INNER_*IN_CH_; i += 256) Wl[i] = W[i];
  __syncthreads();
  const int b = blockIdx.x / (T_/256);
  const int t = (blockIdx.x % (T_/256))*256 + threadIdx.x;
  for (int ob = 0; ob < INNER_; ob += 16) {
    float acc[16];
#pragma unroll
    for (int j = 0; j < 16; ++j) acc[j] = bias[ob+j];
    for (int c = 0; c < IN_CH_; ++c) {
      float xv = x[((size_t)b*IN_CH_ + c)*T_ + t];
#pragma unroll
      for (int j = 0; j < 16; ++j) acc[j] += Wl[(ob+j)*IN_CH_ + c] * xv;
    }
#pragma unroll
    for (int j = 0; j < 16; ++j) h[((size_t)b*INNER_ + ob + j)*T_ + t] = acc[j];
  }
}

// ------------------------------------------------ kernel-predictor front end
__global__ void predictor_kernel(
    const float* __restrict__ c,
    const float* __restrict__ iW, const float* __restrict__ ib,
    const float* __restrict__ r1W, const float* __restrict__ r1b,
    const float* __restrict__ r2W, const float* __restrict__ r2b,
    int n, float* __restrict__ hc, _Float16* __restrict__ hch) {
  const int j = threadIdx.x;            // column = b*L + l
  if (j >= B_*L_) return;
  const int b = j / L_, l = j % L_;
  float h1[KH_], r1[KH_];
  for (int ch = 0; ch < KH_; ++ch) {
    float a = ib[n*KH_ + ch];
    for (int cc = 0; cc < COND_; ++cc) {
      const float* wrow = iW + (((size_t)(n*KH_ + ch))*COND_ + cc)*5;
      const float* crow = c + ((size_t)b*COND_ + cc)*L_;
#pragma unroll
      for (int kk = 0; kk < 5; ++kk) {
        int p = l + kk - 2;
        if (p >= 0 && p < L_) a += wrow[kk] * crow[p];
      }
    }
    h1[ch] = (a >= 0.f) ? a : 0.1f*a;    // leaky relu
  }
  for (int o = 0; o < KH_; ++o) {
    float a = r1b[n*KH_ + o];
    for (int i2 = 0; i2 < KH_; ++i2) a += r1W[((size_t)(n*KH_+o))*KH_ + i2]*h1[i2];
    r1[o] = (a >= 0.f) ? a : 0.1f*a;
  }
  for (int o = 0; o < KH_; ++o) {
    float a = r2b[n*KH_ + o];
    for (int i2 = 0; i2 < KH_; ++i2) a += r2W[((size_t)(n*KH_+o))*KH_ + i2]*r1[i2];
    float v = h1[o] + a;
    hc[(size_t)j*KH_ + o]  = v;
    hch[(size_t)j*KH_ + o] = (_Float16)v;
  }
}

// ------------------------------- predicted-kernel GEMM: kern_W @ hc  (WMMA)
// rows m = (cin*COUT + o)*K + kk ; stored f16 as kerh[col][o][kd'] with
// kd' = kk*128 + cin  (contiguous-cin A layout for the LVC GEMM)
__global__ __launch_bounds__(256) void kernelgen_kernel(
    const float* __restrict__ kern_W, const float* __restrict__ kern_b,
    const _Float16* __restrict__ hch, _Float16* __restrict__ kerh,
    int n, int layer) {
  const int wave = threadIdx.x >> 5;
  const int lane = threadIdx.x & 31;
  const int rlo  = lane & 15;
  const int hi   = lane >> 4;                 // lane half
  const int m0   = (blockIdx.x*8 + wave)*16;
  const size_t wbase = ((size_t)n*KCH_ + (size_t)layer*MROWS_)*KH_;
  const float* arow = kern_W + wbase + (size_t)(m0 + rlo)*KH_;
  // A fragments: rows m0..m0+15, K(ch) split 0..31 / 32..63
  v16h a0, a1;
  {
    const int cb0 = hi*8;
#pragma unroll
    for (int q = 0; q < 8; ++q) { a0[q] = (_Float16)arow[cb0+q];    a0[8+q] = (_Float16)arow[cb0+16+q]; }
    const int cb1 = 32 + hi*8;
#pragma unroll
    for (int q = 0; q < 8; ++q) { a1[q] = (_Float16)arow[cb1+q];    a1[8+q] = (_Float16)arow[cb1+16+q]; }
  }
  for (int ct = 0; ct < 8; ++ct) {            // 8 tiles cover 128 columns
    const int col = ct*16 + rlo;
    const _Float16* bp = hch + (size_t)col*KH_ + hi*8;
    v16h b0, b1;
#pragma unroll
    for (int q = 0; q < 8; ++q) { b0[q] = bp[q]; b0[8+q] = bp[16+q]; b1[q] = bp[32+q]; b1[8+q] = bp[48+q]; }
    v8f acc = {};
    acc = __builtin_amdgcn_wmma_f32_16x16x32_f16(false, a0, false, b0, (short)0, acc, false, false);
    acc = __builtin_amdgcn_wmma_f32_16x16x32_f16(false, a1, false, b1, (short)0, acc, false, false);
#pragma unroll
    for (int r = 0; r < 8; ++r) {
      const int m   = m0 + r + hi*8;          // D: M = r (+8 for high lanes)
      const int cin = m / (COUT_*K_);
      const int rem = m - cin*(COUT_*K_);
      const int o   = rem / K_;
      const int kk  = rem - o*K_;
      const float v = acc[r] + kern_b[(size_t)n*KCH_ + (size_t)layer*MROWS_ + m];
      kerh[(size_t)col*(COUT_*KD_) + (size_t)o*KD_ + kk*INNER_ + cin] = (_Float16)v;
    }
  }
}

// ------------------------------------------------------- per-layer bias head
__global__ void biasgen_kernel(
    const float* __restrict__ bW, const float* __restrict__ bb,
    const float* __restrict__ hc, float* __restrict__ biasbuf,
    int n, int layer) {
  const int idx = blockIdx.x*256 + threadIdx.x;   // o*128 + col
  const int col = idx & 127;
  const int o   = idx >> 7;
  const float* wr  = bW + ((size_t)n*BCH_ + (size_t)layer*COUT_ + o)*KH_;
  const float* hcc = hc + (size_t)col*KH_;
  float a = bb[(size_t)n*BCH_ + (size_t)layer*COUT_ + o];
  for (int i = 0; i < KH_; ++i) a += wr[i]*hcc[i];
  biasbuf[(size_t)col*COUT_ + o] = a;
}

// --------------------------- location-variable conv layer + gated activation
// One workgroup per window (b,l): out[256x256] = ker[256x384] x xshift[384x256]
__global__ __launch_bounds__(256) void lvc_kernel(
    const float* __restrict__ xin, const _Float16* __restrict__ kerh,
    const float* __restrict__ biasbuf, const float* __restrict__ resid,
    float* __restrict__ xout, int dil) {
  extern __shared__ char smem[];
  _Float16* kA = (_Float16*)smem;                            // 256 x KDP_ f16
  _Float16* xs = (_Float16*)(smem + (size_t)COUT_*KDP_*2);   // XROWS_ x XS2_ f16 [pos][cin]
  const int col = blockIdx.x;
  const int b   = col >> 6;
  const int l   = col & 63;
  // ---- stage predicted kernel into (padded) LDS rows: 48 b128 chunks / row
  {
    const _Float16* src = kerh + (size_t)col*(COUT_*KD_);
    for (int e = threadIdx.x; e < COUT_*48; e += 256) {
      const int o = e / 48, q = e - o*48;
#if USE_ASYNC_LDS
      __builtin_amdgcn_global_load_async_to_lds_b128(
          ASYNC_GV4(src + (size_t)o*KD_ + q*8),
          ASYNC_LV4(kA + (size_t)o*KDP_ + q*8), 0, 0);
#else
      *(uint4*)(kA + (size_t)o*KDP_ + q*8) = *(const uint4*)(src + (size_t)o*KD_ + q*8);
#endif
    }
  }
  // ---- stage dilated x window transposed: xs[pos][cin] (f32 -> f16)
  {
    const int wlen = HOP_ + 2*dil;
    const int ws0  = l*HOP_ - dil;
    for (int e = threadIdx.x; e < wlen*INNER_; e += 256) {
      const int pos = e >> 7;
      const int cin = e & 127;
      const int gp  = ws0 + pos;
      const float v = (gp >= 0 && gp < T_) ? xin[((size_t)b*INNER_ + cin)*T_ + gp] : 0.f;
      xs[pos*XS2_ + cin] = (_Float16)v;
    }
  }
#if USE_ASYNC_LDS
  __builtin_amdgcn_s_wait_asynccnt(0);
#endif
  __syncthreads();

  const int wave = threadIdx.x >> 5;
  const int lane = threadIdx.x & 31;
  const int rlo  = lane & 15;
  const int hi   = lane >> 4;
  const int oA   = wave*16;               // each wave: channel pair (oA, oA+128)
  const _Float16* raBase = kA + (size_t)(oA + rlo)*KDP_ + hi*8;
  for (int tp = 0; tp < 8; ++tp) {        // t-tile pairs: A frag reused twice
    v8f acc00 = {}, acc01 = {}, acc10 = {}, acc11 = {};
    const int t0c = tp*32 + rlo;
    const int t1c = t0c + 16;
    for (int ks = 0; ks < 12; ++ks) {     // Kdim = 384 = 12 x 32
      const int kd0  = ks*32;
      const int kk   = kd0 >> 7;          // conv tap (fragment never crosses taps)
      const int cinb = (kd0 & 127) + hi*8;
      // B fragments: cin 8h..8h+7 and 16+8h..16+8h+7 per lane -> 2x ds_load_b128 each
      const _Float16* bp0 = xs + (t0c + kk*dil)*XS2_ + cinb;
      v8h u0 = *(const v8h*)bp0;  v8h u1 = *(const v8h*)(bp0 + 16);
      const _Float16* bp1 = xs + (t1c + kk*dil)*XS2_ + cinb;
      v8h w0 = *(const v8h*)bp1;  v8h w1 = *(const v8h*)(bp1 + 16);
      v16h bf0, bf1;
#pragma unroll
      for (int q = 0; q < 8; ++q) { bf0[q] = u0[q]; bf0[8+q] = u1[q];
                                    bf1[q] = w0[q]; bf1[8+q] = w1[q]; }
      // A fragments for channel rows oA and oA+128
      const _Float16* ra = raBase + kd0;
      v8h p0 = *(const v8h*)(ra);       v8h p1 = *(const v8h*)(ra + 16);
      const _Float16* rb = ra + (size_t)INNER_*KDP_;
      v8h q0 = *(const v8h*)(rb);       v8h q1 = *(const v8h*)(rb + 16);
      v16h af, ag;
#pragma unroll
      for (int q = 0; q < 8; ++q) { af[q] = p0[q]; af[8+q] = p1[q];
                                    ag[q] = q0[q]; ag[8+q] = q1[q]; }
      acc00 = __builtin_amdgcn_wmma_f32_16x16x32_f16(false, af, false, bf0, (short)0, acc00, false, false);
      acc01 = __builtin_amdgcn_wmma_f32_16x16x32_f16(false, af, false, bf1, (short)0, acc01, false, false);
      acc10 = __builtin_amdgcn_wmma_f32_16x16x32_f16(false, ag, false, bf0, (short)0, acc10, false, false);
      acc11 = __builtin_amdgcn_wmma_f32_16x16x32_f16(false, ag, false, bf1, (short)0, acc11, false, false);
    }
#pragma unroll
    for (int r = 0; r < 8; ++r) {         // gate: sigmoid(lo)*tanh(hi)
      const int ch  = oA + r + hi*8;
      const float bl = biasbuf[(size_t)col*COUT_ + ch];
      const float bh = biasbuf[(size_t)col*COUT_ + ch + INNER_];
      const size_t obase = ((size_t)b*INNER_ + ch)*T_ + (size_t)l*HOP_;
      {
        const float lo = acc00[r] + bl, hv = acc10[r] + bh;
        float v = (1.f/(1.f + __expf(-lo))) * tanhf(hv);
        if (resid) v += resid[obase + t0c];
        xout[obase + t0c] = v;
      }
      {
        const float lo = acc01[r] + bl, hv = acc11[r] + bh;
        float v = (1.f/(1.f + __expf(-lo))) * tanhf(hv);
        if (resid) v += resid[obase + t1c];
        xout[obase + t1c] = v;
      }
    }
  }
}

// ------------------------------------------------------------ final 2x conv
__global__ __launch_bounds__(256) void final_kernel(
    const float* __restrict__ h, const float* __restrict__ W1,
    const float* __restrict__ b1, const float* __restrict__ W2,
    const float* __restrict__ b2, float* __restrict__ out) {
  extern __shared__ char smem[];
  float* hs = (float*)smem;               // 128 x 256
  const int b  = blockIdx.x / (T_/256);
  const int t0 = (blockIdx.x % (T_/256))*256;
  for (int e = threadIdx.x; e < INNER_*256; e += 256) {
    const int i = e >> 8, tp = e & 255;
    hs[e] = fmaxf(h[((size_t)b*INNER_ + i)*T_ + t0 + tp], 0.f);
  }
  __syncthreads();
  float acc2 = b2[0];
  for (int o = 0; o < INNER_; ++o) {
    float a = b1[o];
    for (int i = 0; i < INNER_; ++i) a += W1[o*INNER_ + i]*hs[i*256 + threadIdx.x];
    acc2 += W2[o]*fmaxf(a, 0.f);
  }
  out[(size_t)b*T_ + t0 + threadIdx.x] = acc2;
}

// ---------------------------------------------------------------- dispatcher
extern "C" void kernel_launch(void* const* d_in, const int* in_sizes, int n_in,
                              void* d_out, int out_size, void* d_ws, size_t ws_size,
                              hipStream_t stream) {
  (void)in_sizes; (void)n_in; (void)out_size; (void)ws_size;
  const float* x      = (const float*)d_in[0];
  const float* c      = (const float*)d_in[1];
  const float* fc_W   = (const float*)d_in[2];
  const float* fc_b   = (const float*)d_in[3];
  const float* inp_W  = (const float*)d_in[4];
  const float* inp_b  = (const float*)d_in[5];
  const float* res1_W = (const float*)d_in[6];
  const float* res1_b = (const float*)d_in[7];
  const float* res2_W = (const float*)d_in[8];
  const float* res2_b = (const float*)d_in[9];
  const float* kern_W = (const float*)d_in[10];
  const float* kern_b = (const float*)d_in[11];
  const float* biasc_W= (const float*)d_in[12];
  const float* biasc_b= (const float*)d_in[13];
  const float* lc1_W  = (const float*)d_in[14];
  const float* lc1_b  = (const float*)d_in[15];
  const float* lc2_W  = (const float*)d_in[16];
  const float* lc2_b  = (const float*)d_in[17];
  float* outp = (float*)d_out;

  char* ws = (char*)d_ws;
  const size_t HS = (size_t)B_*INNER_*T_*sizeof(float);      // 16 MB per activation buf
  float*    H0   = (float*)ws;
  float*    PA   = (float*)(ws + HS);
  float*    PB   = (float*)(ws + 2*HS);
  _Float16* kerh = (_Float16*)(ws + 3*HS);                   // 25.2 MB
  size_t off = 3*HS + (size_t)B_*L_*COUT_*KD_*sizeof(_Float16);
  float*    hc   = (float*)(ws + off);    off += (size_t)B_*L_*KH_*sizeof(float);
  _Float16* hch  = (_Float16*)(ws + off); off += (size_t)B_*L_*KH_*sizeof(_Float16);
  float* biasbuf = (float*)(ws + off);

  fc_kernel<<<B_*(T_/256), 256, 0, stream>>>(x, fc_W, fc_b, H0);

  const size_t lvcShm = (size_t)COUT_*KDP_*2 + (size_t)XROWS_*XS2_*2;  // ~273 KB

  for (int n = 0; n < NBLOCK_; ++n) {
    predictor_kernel<<<1, 128, 0, stream>>>(c, inp_W, inp_b, res1_W, res1_b,
                                            res2_W, res2_b, n, hc, hch);
    float* cur = H0;
    for (int i = 0; i < LAYERS_; ++i) {
      kernelgen_kernel<<<MROWS_/(16*8), 256, 0, stream>>>(kern_W, kern_b, hch, kerh, n, i);
      biasgen_kernel<<<(COUT_*B_*L_)/256, 256, 0, stream>>>(biasc_W, biasc_b, hc, biasbuf, n, i);
      float* nxt;
      const float* resid = nullptr;
      if (i < LAYERS_-1) {
        nxt = (cur == PA) ? PB : PA;
      } else if (n == 0) {
        nxt = H0;                 // block-0 output becomes block-1 input
      } else {
        nxt = PA;                 // block-1 final, residual add with H0
        resid = H0;
      }
      lvc_kernel<<<B_*L_, 256, lvcShm, stream>>>(cur, kerh, biasbuf, resid, nxt, 1 << i);
      cur = nxt;
    }
  }
  // final activations live in PA (block1, layer4 target)
  final_kernel<<<B_*(T_/256), 256, (size_t)INNER_*256*sizeof(float), stream>>>(
      PA, lc1_W, lc1_b, lc2_W, lc2_b, outp);
}
